// MultiNodeAggregation_29841432773230
// MI455X (gfx1250) — matrix operand order repaired and checked by reading, not compiled, not run on hardware
//
#include <hip/hip_runtime.h>

typedef __attribute__((ext_vector_type(16))) _Float16 v16h;
typedef __attribute__((ext_vector_type(8)))  _Float16 v8h;
typedef __attribute__((ext_vector_type(8)))  float    v8f;

#define HDIM   256
#define NNODES 131072
#define NBATCH 128
#define LDSTR  264   // padded f16 row stride in LDS (16B aligned, conflict-spread)
#define SPLIT  8     // blocks per segment in the aggregation pass

// Hardware tanh (V_TANH_F32 on gfx1250) when the builtin exists.
#if __has_builtin(__builtin_amdgcn_tanhf)
#define FAST_TANH(x) __builtin_amdgcn_tanhf(x)
#elif __has_builtin(__builtin_amdgcn_tanh_f32)
#define FAST_TANH(x) __builtin_amdgcn_tanh_f32(x)
#else
#define FAST_TANH(x) tanhf(x)
#endif

// ---------------- Kernel 0: W1 (HxH f32, [k][j]) -> W1t (HxH f16, [j][k]) ----------------
__global__ void w1_transpose_f16_kernel(const float* __restrict__ W1,
                                        _Float16* __restrict__ W1t) {
    int idx = blockIdx.x * blockDim.x + threadIdx.x;   // = j*256 + k
    int j = idx >> 8;
    int k = idx & 255;
    W1t[idx] = (_Float16)W1[k * HDIM + j];             // coalesced f16 writes
}

// B-fragment load for tile T: all displacements are immediates in the 16-bit DS
// offset field (tiles 0..7 off p0, tiles 8..15 off p1 — p1 kept opaque by caller).
template <int T>
static __device__ inline v16h load_b(const _Float16* p0, const _Float16* p1) {
    const _Float16* bp = (T < 8) ? (p0 + T * 16 * LDSTR)
                                 : (p1 + (T - 8) * 16 * LDSTR);
    union { v16h v; v8h h[2]; } bu;
    bu.h[0] = *(const v8h*)(bp);
    bu.h[1] = *(const v8h*)(bp + 16);
    return bu.v;
}

// Double-buffered WMMA chain: issue tile T+1's B loads BEFORE tile T's WMMA so each
// WMMA only needs dscnt<=2 (one DS round-trip in flight) instead of dscnt==0.
// reuse_a is set on tiles 0..14 (next WMMA is identical and reuses the A fragment).
template <int T>
struct WmmaChain {
    static __device__ inline void run(const v16h& a, const _Float16* p0,
                                      const _Float16* p1, v8f* acc, const v16h& bcur) {
        v16h bnext = load_b<T + 1>(p0, p1);
        acc[T] = __builtin_amdgcn_wmma_f32_16x16x32_f16(
            false, a, false, bcur, (short)0, acc[T], true, false);
        WmmaChain<T + 1>::run(a, p0, p1, acc, bnext);
    }
};
template <>
struct WmmaChain<15> {
    static __device__ inline void run(const v16h& a, const _Float16* p0,
                                      const _Float16* p1, v8f* acc, const v16h& bcur) {
        acc[15] = __builtin_amdgcn_wmma_f32_16x16x32_f16(
            false, a, false, bcur, (short)0, acc[15], false, false);
    }
};

// ---------------- Kernel 1: scores[n] = tanh(X[n,:]W1 + b1) . W2  (WMMA) ----------------
__global__ __launch_bounds__(256, 1)
void score_wmma_kernel(const float* __restrict__ X,
                       const _Float16* __restrict__ W1t,
                       const float* __restrict__ b1,
                       const float* __restrict__ W2,
                       float* __restrict__ scores) {
    extern __shared__ _Float16 sW[];   // [256][LDSTR] : W1t rows (j-major, k contiguous)

    const int tid = threadIdx.x;

    // Stage W1t into LDS as 16B chunks (row r has 32 chunks of 8 f16)
    {
        const uint4* src = (const uint4*)W1t;
        for (int c = tid; c < 256 * 32; c += 256) {
            int r = c >> 5, s = c & 31;
            *(uint4*)(&sW[r * LDSTR + s * 8]) = src[r * 32 + s];
        }
    }
    __syncthreads();

    const int lane = tid & 31;
    const int wave = tid >> 5;
    const int lo   = lane & 15;
    const int hi   = lane >> 4;          // 0: lanes 0-15, 1: lanes 16-31
    const int m0   = blockIdx.x * 128 + wave * 16;   // 16 rows per wave
    const float* Xrow = X + (size_t)(m0 + lo) * HDIM;

    // Per-lane B base: row j = lo (within each 16-row tile), K {0..7,16..23} (hi=0) / {8..15,24..31} (hi=1).
    const _Float16* bbase = sW + lo * LDSTR + hi * 8;

    v8f acc[16] = {};                    // 16 N-tiles covering all 256 columns

    for (int k0 = 0; k0 < 8; ++k0) {     // K in steps of 32
        // A fragment (16x32 f16): lo lanes K {0..7,16..23}, hi lanes K {8..15,24..31}
        const float* p = Xrow + k0 * 32 + hi * 8;
        float4 f0 = *(const float4*)(p);
        float4 f1 = *(const float4*)(p + 4);
        float4 f2 = *(const float4*)(p + 16);
        float4 f3 = *(const float4*)(p + 20);
        v16h a;
        a[0]=(_Float16)f0.x; a[1]=(_Float16)f0.y; a[2]=(_Float16)f0.z; a[3]=(_Float16)f0.w;
        a[4]=(_Float16)f1.x; a[5]=(_Float16)f1.y; a[6]=(_Float16)f1.z; a[7]=(_Float16)f1.w;
        a[8]=(_Float16)f2.x; a[9]=(_Float16)f2.y; a[10]=(_Float16)f2.z; a[11]=(_Float16)f2.w;
        a[12]=(_Float16)f3.x; a[13]=(_Float16)f3.y; a[14]=(_Float16)f3.z; a[15]=(_Float16)f3.w;

        const _Float16* p0 = bbase + k0 * 32;          // tiles 0..7  (imm offsets)
        // Opaque element-offset so the compiler keeps a SECOND DS base register
        // instead of folding tiles 8..15 into p0 + >64KB constants.
        unsigned off1 = (unsigned)(k0 * 32) + 8u * 16u * LDSTR;
        asm volatile("" : "+v"(off1));
        const _Float16* p1 = bbase + off1;             // tiles 8..15 (imm offsets)

        v16h b0 = load_b<0>(p0, p1);                   // prime the double buffer
        WmmaChain<0>::run(a, p0, p1, acc, b0);
    }

    // Fused epilogue: tanh + dot(W2) + reduce over columns
    float part[8];
    #pragma unroll
    for (int r = 0; r < 8; ++r) part[r] = 0.f;

    #pragma unroll
    for (int t = 0; t < 16; ++t) {
        int j = t * 16 + lo;
        float bj = b1[j];
        float wj = W2[j];
        #pragma unroll
        for (int r = 0; r < 8; ++r)
            part[r] += FAST_TANH(acc[t][r] + bj) * wj;   // elem (m0 + r + hi*8, j)
    }

    #pragma unroll
    for (int r = 0; r < 8; ++r) {
        float v = part[r];
        v += __shfl_xor(v, 1, 32);
        v += __shfl_xor(v, 2, 32);
        v += __shfl_xor(v, 4, 32);
        v += __shfl_xor(v, 8, 32);       // butterfly within each 16-lane half
        if (lo == 0) scores[m0 + hi * 8 + r] = v;
    }
}

// ---------------- Kernel 2: per-segment softmax stats (max, 1/sum-exp) ----------------
__global__ __launch_bounds__(256)
void seg_stats_kernel(const int* __restrict__ off,
                      const float* __restrict__ scores,
                      float* __restrict__ stats) {   // stats[2*i] = max, stats[2*i+1] = invZ
    __shared__ float red[256];
    const int i = blockIdx.x;
    const int tid = threadIdx.x;

    // Bounds exactly per _segment_bounds:
    //   starts = [off[0], off[0..B-2]] ; ends = [off[2..B-1], N, N]
    const int start = (i == 0) ? off[0] : off[i - 1];
    const int end   = (i < NBATCH - 2) ? off[i + 2] : NNODES;

    float mx = -3.402823466e38f;
    for (int n = start + tid; n < end; n += 256) mx = fmaxf(mx, scores[n]);
    red[tid] = mx; __syncthreads();
    for (int s = 128; s > 0; s >>= 1) {
        if (tid < s) red[tid] = fmaxf(red[tid], red[tid + s]);
        __syncthreads();
    }
    mx = red[0]; __syncthreads();

    float sm = 0.f;
    for (int n = start + tid; n < end; n += 256) sm += __expf(scores[n] - mx);
    red[tid] = sm; __syncthreads();
    for (int s = 128; s > 0; s >>= 1) {
        if (tid < s) red[tid] += red[tid + s];
        __syncthreads();
    }
    if (tid == 0) {
        stats[2 * i]     = mx;
        stats[2 * i + 1] = 1.f / red[0];
    }
}

// ---------------- Kernel 3: sliced weighted aggregation into partials ----------------
__global__ __launch_bounds__(256)
void seg_partial_kernel(const float* __restrict__ X,
                        const int* __restrict__ off,
                        const float* __restrict__ scores,
                        const float* __restrict__ stats,
                        float* __restrict__ partials) {  // [NBATCH*SPLIT][HDIM]
    __shared__ float wch[256];
    __shared__ float4 accb[4][64];

    const int bid = blockIdx.x;
    const int i = bid >> 3;              // segment
    const int q = bid & (SPLIT - 1);     // slice within segment
    const int tid = threadIdx.x;

    const int start = (i == 0) ? off[0] : off[i - 1];
    const int end   = (i < NBATCH - 2) ? off[i + 2] : NNODES;
    const int L  = end - start;
    const int qs = start + (L * q) / SPLIT;
    const int qe = start + (L * (q + 1)) / SPLIT;

    const float mx   = stats[2 * i];
    const float invZ = stats[2 * i + 1];

    const int g = tid >> 6;              // node sub-group 0..3
    const int c = tid & 63;              // column group: cols 4c..4c+3
    float4 acc = make_float4(0.f, 0.f, 0.f, 0.f);

    for (int base = qs; base < qe; base += 256) {
        int n = base + tid;
        wch[tid] = (n < qe) ? __expf(scores[n] - mx) * invZ : 0.f;
        __syncthreads();
        const int cnt = min(256, qe - base);
        for (int jj = g; jj < cnt; jj += 4) {
            float w = wch[jj];           // LDS broadcast within group
            const float4 x = *(const float4*)(X + (size_t)(base + jj) * HDIM + c * 4);
            acc.x += w * x.x; acc.y += w * x.y; acc.z += w * x.z; acc.w += w * x.w;
        }
        __syncthreads();
    }

    accb[g][c] = acc;
    __syncthreads();
    if (g == 0) {
        float4 r0 = accb[0][c], r1 = accb[1][c], r2 = accb[2][c], r3 = accb[3][c];
        float4 r;
        r.x = r0.x + r1.x + r2.x + r3.x;
        r.y = r0.y + r1.y + r2.y + r3.y;
        r.z = r0.z + r1.z + r2.z + r3.z;
        r.w = r0.w + r1.w + r2.w + r3.w;
        *(float4*)(partials + (size_t)bid * HDIM + c * 4) = r;
    }
}

// ---------------- Kernel 4: deterministic final reduction over slices ----------------
__global__ __launch_bounds__(256)
void final_reduce_kernel(const float* __restrict__ partials,
                         float* __restrict__ out) {
    const int i = blockIdx.x;
    const int tid = threadIdx.x;
    float s = 0.f;
    #pragma unroll
    for (int q = 0; q < SPLIT; ++q)
        s += partials[((size_t)i * SPLIT + q) * HDIM + tid];
    out[(size_t)i * HDIM + tid] = s;
}

// ---------------- Launch ----------------
extern "C" void kernel_launch(void* const* d_in, const int* in_sizes, int n_in,
                              void* d_out, int out_size, void* d_ws, size_t ws_size,
                              hipStream_t stream) {
    const float* X   = (const float*)d_in[0];   // node_features (N, H)
    const int*   off = (const int*)d_in[1];     // offsets (B,)
    const float* W1  = (const float*)d_in[2];   // (H, H)
    const float* b1  = (const float*)d_in[3];   // (H,)
    const float* W2  = (const float*)d_in[4];   // (H, 1)
    // d_in[5] = b2: constant shift, cancels in softmax — unused.
    float* out = (float*)d_out;

    char* ws = (char*)d_ws;
    float*    scores   = (float*)ws;                                   // N floats (512KB)
    _Float16* W1t      = (_Float16*)(ws + NNODES * sizeof(float));     // 128KB
    float*    stats    = (float*)(ws + NNODES * sizeof(float) + HDIM * HDIM * sizeof(_Float16));
    float*    partials = stats + 2 * NBATCH;                           // NBATCH*SPLIT*HDIM floats

    w1_transpose_f16_kernel<<<(HDIM * HDIM) / 256, 256, 0, stream>>>(W1, W1t);

    score_wmma_kernel<<<NNODES / 128, 256, LDSTR * 256 * sizeof(_Float16), stream>>>(
        X, W1t, b1, W2, scores);

    seg_stats_kernel<<<NBATCH, 256, 0, stream>>>(off, scores, stats);

    seg_partial_kernel<<<NBATCH * SPLIT, 256, 0, stream>>>(X, off, scores, stats, partials);

    final_reduce_kernel<<<NBATCH, 256, 0, stream>>>(partials, out);
}
